// NGPTCausalAttention_64424509440421
// MI455X (gfx1250) — matrix operand order repaired and checked by reading, not compiled, not run on hardware
//
#include <hip/hip_runtime.h>
#include <hip/hip_bf16.h>
#include <math.h>

// ---------------------------------------------------------------------------
// Types for CDNA5 WMMA (wave32): v_wmma_f32_16x16x32_bf16
// ---------------------------------------------------------------------------
typedef __attribute__((ext_vector_type(16))) __bf16 v16bf;
typedef __attribute__((ext_vector_type(8)))  float  v8f;

#define WARP 32

__device__ __forceinline__ unsigned short f2bf(float f) {
  union { float f; unsigned u; } v; v.f = f;
  unsigned r = v.u + 0x7FFFu + ((v.u >> 16) & 1u); // round-to-nearest-even
  return (unsigned short)(r >> 16);
}

__device__ __forceinline__ v8f zero8() {
  v8f z;
#pragma unroll
  for (int i = 0; i < 8; ++i) z[i] = 0.0f;
  return z;
}

// Async global->LDS copy of 16 bytes per lane (CDNA5 async engine, ASYNCcnt).
// lds operand = byte offset within LDS (low 32 bits of generic shared ptr).
__device__ __forceinline__ void async_b128(unsigned short* ldsp,
                                           const unsigned short* gp) {
  unsigned lds_off = (unsigned)(uintptr_t)ldsp;
  asm volatile("global_load_async_to_lds_b128 %0, %1, off"
               :: "v"(lds_off), "v"(gp) : "memory");
}
#define WAIT_ASYNC_6() asm volatile("s_wait_asynccnt 0x6" ::: "memory")
#define WAIT_ASYNC_0() asm volatile("s_wait_asynccnt 0x0" ::: "memory")

union Frag { v16bf v; uint4 q[2]; };

// A-fragment (16x32 bf16, MxK): lane m=lane%16; half h=lane/16.
// VGPR0..3 hold K = k0 + 8h + 0..7 ; VGPR4..7 hold K = k0 + 16 + 8h + 0..7
__device__ __forceinline__ v16bf load_afrag(const unsigned short* lds, int row0,
                                            int k0, int stride, int lane) {
  int m = lane & 15, h = lane >> 4;
  const unsigned short* p = lds + (row0 + m) * stride + k0 + h * 8;
  Frag f;
  f.q[0] = *(const uint4*)(p);
  f.q[1] = *(const uint4*)(p + 16);
  return f.v;
}

// B-fragment (32x16 bf16, KxN): lane column n=lane%16; half h=lane/16 holds
// K = k0 + 8h + 0..7 and k0 + 16 + 8h + 0..7. lds layout must be [n][k] (k contiguous).
__device__ __forceinline__ v16bf load_bfrag(const unsigned short* lds, int n0,
                                            int k0, int stride, int lane) {
  int n = lane & 15, h = lane >> 4;
  const unsigned short* p = lds + (n0 + n) * stride + k0 + h * 8;
  Frag f;
  f.q[0] = *(const uint4*)(p);
  f.q[1] = *(const uint4*)(p + 16);
  return f.v;
}

__device__ __forceinline__ v8f wmma_bf16(v16bf a, v16bf b, v8f c) {
  return __builtin_amdgcn_wmma_f32_16x16x32_bf16(false, a, false, b,
                                                 (short)0, c, false, false);
}

// ---------------------------------------------------------------------------
// Prep kernels: f32 -> bf16 (row-major), and f32 -> bf16 transposed [n][k]
// ---------------------------------------------------------------------------
__global__ void k_f2bf(const float* __restrict__ s, unsigned short* __restrict__ d, int n) {
  for (int i = blockIdx.x * blockDim.x + threadIdx.x; i < n; i += gridDim.x * blockDim.x)
    d[i] = f2bf(s[i]);
}

__global__ void k_transpose_bf(const float* __restrict__ w, unsigned short* __restrict__ wT,
                               int K, int N) {
  for (int i = blockIdx.x * blockDim.x + threadIdx.x; i < K * N; i += gridDim.x * blockDim.x) {
    int k = i / N, n = i - k * N;
    wT[(size_t)n * K + k] = f2bf(w[i]);
  }
}

// ---------------------------------------------------------------------------
// Shared GEMM tile: C(64x128) = A(64xK) * B(KxN-slice), A row-major bf16,
// Bt = B^T row-major bf16 ([n][k]).  Block = 128 threads = 4 waves.
// Wave tile M=32 x N=64: wave w -> row-half mh=w>>1, col-half nh=w&1.
// Col-half nh owns N-subtiles {2nh, 2nh+1, 2nh+4, 2nh+5} so that RoPE pairs
// (tile t, t+4 i.e. col j, j+64) stay within one wave.
// Per K-step per wave: 2 A-frags + 4 B-frags (12 ds_load_b128) -> 8 WMMA.
// acc[8] = mi*4 + j (mi: 16-row subtile, j: index into the wave's tile set).
// Double-buffered: async global->LDS loads for stage k+1 overlap WMMA on k.
// Per thread per stage: 2 async b128 (A) + 4 async b128 (B) = 6 outstanding.
// ---------------------------------------------------------------------------
__device__ __forceinline__ int wave_tile(int nh, int j) {
  return nh * 2 + (j & 1) + (j & 2) * 2;   // {2nh, 2nh+1, 2nh+4, 2nh+5}
}

__device__ __forceinline__ void gemm_issue_stage(
    const unsigned short* __restrict__ A, const unsigned short* __restrict__ Bt,
    int lda, int ldb, int mbase, int nbase, int k0,
    unsigned short* Alds, unsigned short* Blds, int st) {
  const int tid = threadIdx.x;
  const int ar = tid >> 1, ap = tid & 1;
  unsigned short* al = Alds + st * (64 * 32) + ar * 32 + ap * 16;
  const unsigned short* ag = A + (size_t)(mbase + ar) * lda + k0 + ap * 16;
  async_b128(al, ag);
  async_b128(al + 8, ag + 8);
  unsigned short* bl = Blds + st * (128 * 32) + tid * 32;
  const unsigned short* bg = Bt + (size_t)(nbase + tid) * ldb + k0;
#pragma unroll
  for (int u = 0; u < 4; ++u) async_b128(bl + u * 8, bg + u * 8);
}

__device__ __forceinline__ void gemm_tile_64x128(
    const unsigned short* __restrict__ A, const unsigned short* __restrict__ Bt,
    int lda, int ldb, int mbase, int nbase, int K,
    unsigned short* Alds, unsigned short* Blds, v8f* acc /* [8]: mi*4+j */) {
  const int tid = threadIdx.x;
  const int wave = tid >> 5, lane = tid & 31;
  const int mh = wave >> 1, nh = wave & 1;

  gemm_issue_stage(A, Bt, lda, ldb, mbase, nbase, 0, Alds, Blds, 0);

  for (int k0 = 0; k0 < K; k0 += 32) {
    const int cur = (k0 >> 5) & 1;
    if (k0 + 32 < K) {
      gemm_issue_stage(A, Bt, lda, ldb, mbase, nbase, k0 + 32, Alds, Blds, cur ^ 1);
      WAIT_ASYNC_6();   // current stage retired; next stage's 6 still in flight
    } else {
      WAIT_ASYNC_0();
    }
    __syncthreads();

    const unsigned short* Ac = Alds + cur * (64 * 32);
    const unsigned short* Bc = Blds + cur * (128 * 32);
    v16bf a0 = load_afrag(Ac, mh * 32, 0, 32, lane);
    v16bf a1 = load_afrag(Ac, mh * 32 + 16, 0, 32, lane);
#pragma unroll
    for (int g = 0; g < 2; ++g) {    // two groups of 2 B-fragments
      v16bf bfr[2];
#pragma unroll
      for (int j2 = 0; j2 < 2; ++j2)
        bfr[j2] = load_bfrag(Bc, wave_tile(nh, g * 2 + j2) * 16, 0, 32, lane);
#pragma unroll
      for (int j2 = 0; j2 < 2; ++j2) {
        acc[g * 2 + j2]     = wmma_bf16(a0, bfr[j2], acc[g * 2 + j2]);
        acc[4 + g * 2 + j2] = wmma_bf16(a1, bfr[j2], acc[4 + g * 2 + j2]);
      }
    }
    __syncthreads();
  }
}

// ---------------------------------------------------------------------------
// Kernel 2: QKV GEMM + RoPE + L2-normalize + s_qk, scatter to [B,H,L,128] bf16
// grid = (4096/64, 48) ; block = 128
// ---------------------------------------------------------------------------
__global__ __launch_bounds__(128, 1) void k_qkv(
    const unsigned short* __restrict__ xb,     // [4096][2048] bf16
    const unsigned short* __restrict__ wqkvT,  // [6144][2048] bf16
    const float* __restrict__ s_qk,
    unsigned short* __restrict__ qb, unsigned short* __restrict__ kb,
    unsigned short* __restrict__ vb) {
  __shared__ __align__(16) unsigned short Alds[2 * 64 * 32];
  __shared__ __align__(16) unsigned short Blds[2 * 128 * 32];
  __shared__ float NormBuf[2][64];   // [col-half][tile row]: partial sum-sq

  const int mblk = blockIdx.x;         // 64-row block of B*L
  const int ch   = blockIdx.y;         // 0..47 head-chunk of 3*D axis
  const int which = ch >> 4;           // 0=q 1=k 2=v
  const int h     = ch & 15;

  v8f acc[8];   // mi*4 + j
#pragma unroll
  for (int i = 0; i < 8; ++i) acc[i] = zero8();

  gemm_tile_64x128(xb, wqkvT, 2048, 2048, mblk * 64, ch * 128, 2048, Alds, Blds, acc);

  const int lane = threadIdx.x & 31, wave = threadIdx.x >> 5;
  const int mh = wave >> 1, nh = wave & 1;
  const int n16 = lane & 15, hf = lane >> 4;
  const float sq = *s_qk;

  if (which < 2) {
    // RoPE: pair j (tiles 2nh,2nh+1 -> cols<64) with j+2 (tiles +4 -> cols+64)
#pragma unroll
    for (int mi = 0; mi < 2; ++mi)
#pragma unroll
      for (int i = 0; i < 8; ++i) {
        int row = mblk * 64 + mh * 32 + mi * 16 + i + 8 * hf;
        float pos = (float)(row & 2047);
#pragma unroll
        for (int j = 0; j < 2; ++j) {
          int jc = wave_tile(nh, j) * 16 + n16;            // 0..63
          float ang = pos * __expf(-(float)jc * 0.14391157f); // ln(1e4)/64
          float sn, cs;
          sincosf(ang, &sn, &cs);
          float lo = acc[mi * 4 + j][i], hi = acc[mi * 4 + j + 2][i];
          acc[mi * 4 + j][i]     = lo * cs - hi * sn;
          acc[mi * 4 + j + 2][i] = hi * cs + lo * sn;
        }
      }
    // partial sum-sq per row over this wave's 64 cols; combine halves via LDS
#pragma unroll
    for (int mi = 0; mi < 2; ++mi)
#pragma unroll
      for (int i = 0; i < 8; ++i) {
        float ss = 0.0f;
#pragma unroll
        for (int j = 0; j < 4; ++j) ss += acc[mi * 4 + j][i] * acc[mi * 4 + j][i];
        for (int off = 1; off < 16; off <<= 1) ss += __shfl_xor(ss, off, WARP);
        if (n16 == 0) NormBuf[nh][mh * 32 + mi * 16 + i + 8 * hf] = ss;
      }
    __syncthreads();
#pragma unroll
    for (int mi = 0; mi < 2; ++mi)
#pragma unroll
      for (int i = 0; i < 8; ++i) {
        int rowl = mh * 32 + mi * 16 + i + 8 * hf;
        float ss = NormBuf[0][rowl] + NormBuf[1][rowl];
        float sc = 1.0f / fmaxf(sqrtf(ss), 1e-12f);
        if (which == 0) sc *= sq;
#pragma unroll
        for (int j = 0; j < 4; ++j) acc[mi * 4 + j][i] *= sc;
      }
  }

  unsigned short* dst = (which == 0) ? qb : (which == 1) ? kb : vb;
#pragma unroll
  for (int mi = 0; mi < 2; ++mi)
#pragma unroll
    for (int j = 0; j < 4; ++j)
#pragma unroll
      for (int i = 0; i < 8; ++i) {
        int row = mblk * 64 + mh * 32 + mi * 16 + i + 8 * hf;
        int b = row >> 11, l = row & 2047;
        int col = wave_tile(nh, j) * 16 + n16;
        dst[((size_t)(b * 16 + h) * 2048 + l) * 128 + col] = f2bf(acc[mi * 4 + j][i]);
      }
}

// ---------------------------------------------------------------------------
// Kernel 3: causal flash attention per (b,h,q-block of 64). block=128 (4 waves)
// grid = (B*H=32, L/64=32)
// ---------------------------------------------------------------------------
__global__ __launch_bounds__(128, 1) void k_attn(
    const unsigned short* __restrict__ qb, const unsigned short* __restrict__ kb,
    const unsigned short* __restrict__ vb, unsigned short* __restrict__ attnb) {
  __shared__ __align__(16) unsigned short Qlds[64 * 128];
  __shared__ __align__(16) unsigned short Klds[64 * 128];
  __shared__ __align__(16) unsigned short VT[128 * 64];    // V transposed [c][kr]
  __shared__ __align__(16) unsigned short Plds[4][16 * 64];

  const int bh = blockIdx.x;           // b*16 + h
  const int b = bh >> 4, h = bh & 15;
  const int qblk = blockIdx.y;
  const int tid = threadIdx.x, wave = tid >> 5, lane = tid & 31;
  const int n16 = lane & 15, hf = lane >> 4;
  const int r = tid >> 1, half = tid & 1;   // staging: row, 64-col half

  { // load Q tile 64x128 via async engine
    const unsigned short* src = qb + ((size_t)bh * 2048 + qblk * 64 + r) * 128 + half * 64;
    unsigned short* dstp = Qlds + r * 128 + half * 64;
#pragma unroll
    for (int u = 0; u < 8; ++u) async_b128(dstp + u * 8, src + u * 8);
    WAIT_ASYNC_0();
  }
  __syncthreads();

  v8f o[8];
#pragma unroll
  for (int i = 0; i < 8; ++i) o[i] = zero8();
  float mi[8], li[8];
#pragma unroll
  for (int i = 0; i < 8; ++i) { mi[i] = -3.0e38f; li[i] = 0.0f; }

  for (int kt = 0; kt <= qblk; ++kt) {
    { // K tile 64x128 row-major via async engine
      const unsigned short* src = kb + ((size_t)bh * 2048 + kt * 64 + r) * 128 + half * 64;
      unsigned short* dstp = Klds + r * 128 + half * 64;
#pragma unroll
      for (int u = 0; u < 8; ++u) async_b128(dstp + u * 8, src + u * 8);
    }
    { // V tile, store transposed: VT[c][kr] (scatter; regular loads)
      const uint4* src = (const uint4*)(vb + ((size_t)bh * 2048 + kt * 64 + r) * 128 + half * 64);
#pragma unroll
      for (int u = 0; u < 8; ++u) {
        uint4 d = src[u];
        const unsigned short* e = (const unsigned short*)&d;
#pragma unroll
        for (int j = 0; j < 8; ++j) {
          int c = half * 64 + u * 8 + j;
          VT[c * 64 + r] = e[j];
        }
      }
    }
    WAIT_ASYNC_0();
    __syncthreads();

    // S = Q * K^T : 16x64 strip per wave (Q A-frag reloaded per kc: saves regs)
    v8f s[4];
#pragma unroll
    for (int nt = 0; nt < 4; ++nt) s[nt] = zero8();
#pragma unroll
    for (int kc = 0; kc < 4; ++kc) {
      v16bf a = load_afrag(Qlds, wave * 16, kc * 32, 128, lane);
      v16bf kf[4];
#pragma unroll
      for (int nt = 0; nt < 4; ++nt) kf[nt] = load_bfrag(Klds, nt * 16, kc * 32, 128, lane);
#pragma unroll
      for (int nt = 0; nt < 4; ++nt) s[nt] = wmma_bf16(a, kf[nt], s[nt]);
    }

    if (kt == qblk) { // causal mask on diagonal block
#pragma unroll
      for (int nt = 0; nt < 4; ++nt)
#pragma unroll
        for (int i = 0; i < 8; ++i) {
          int ql = wave * 16 + i + 8 * hf;
          int kl = nt * 16 + n16;
          if (kl > ql) s[nt][i] = -3.0e38f;
        }
    }

    // online softmax (row stats per accumulator element, 16-lane reduce)
    float mx[8];
#pragma unroll
    for (int i = 0; i < 8; ++i) {
      float m = s[0][i];
      m = fmaxf(m, s[1][i]); m = fmaxf(m, s[2][i]); m = fmaxf(m, s[3][i]);
      for (int off = 1; off < 16; off <<= 1) m = fmaxf(m, __shfl_xor(m, off, WARP));
      mx[i] = m;
    }
#pragma unroll
    for (int i = 0; i < 8; ++i) {
      float nm = fmaxf(mi[i], mx[i]);
      float al = __expf(mi[i] - nm);
      mi[i] = nm;
      li[i] *= al;
#pragma unroll
      for (int nt = 0; nt < 8; ++nt) o[nt][i] *= al;
    }
    float rs[8];
#pragma unroll
    for (int i = 0; i < 8; ++i) rs[i] = 0.0f;
#pragma unroll
    for (int nt = 0; nt < 4; ++nt)
#pragma unroll
      for (int i = 0; i < 8; ++i) {
        float p = __expf(s[nt][i] - mi[i]);
        rs[i] += p;
        Plds[wave][(i + 8 * hf) * 64 + nt * 16 + n16] = f2bf(p);
      }
#pragma unroll
    for (int i = 0; i < 8; ++i) {
      for (int off = 1; off < 16; off <<= 1) rs[i] += __shfl_xor(rs[i], off, WARP);
      li[i] += rs[i];
    }

    asm volatile("s_wait_dscnt 0" ::: "memory"); // P stores (wave-local) visible

    // O += P * V  (V fragments in groups of 4 to bound register pressure)
#pragma unroll
    for (int kc2 = 0; kc2 < 2; ++kc2) {
      v16bf pa = load_afrag(Plds[wave], 0, kc2 * 32, 64, lane);
#pragma unroll
      for (int g = 0; g < 2; ++g) {
        v16bf vf[4];
#pragma unroll
        for (int j = 0; j < 4; ++j) vf[j] = load_bfrag(VT, (g * 4 + j) * 16, kc2 * 32, 64, lane);
#pragma unroll
        for (int j = 0; j < 4; ++j) o[g * 4 + j] = wmma_bf16(pa, vf[j], o[g * 4 + j]);
      }
    }
    __syncthreads();
  }

  // finalize and scatter to attn buffer [B][L][D] bf16
#pragma unroll
  for (int i = 0; i < 8; ++i) {
    float inv = 1.0f / li[i];
    int l_abs = qblk * 64 + wave * 16 + i + 8 * hf;
#pragma unroll
    for (int nt = 0; nt < 8; ++nt) {
      int col = h * 128 + nt * 16 + n16;
      attnb[((size_t)(b * 2048 + l_abs)) * 2048 + col] = f2bf(o[nt][i] * inv);
    }
  }
}

// ---------------------------------------------------------------------------
// Kernel 4: out projection, f32 output. grid = (4096/64, 2048/128), block=128
// ---------------------------------------------------------------------------
__global__ __launch_bounds__(128, 1) void k_outproj(
    const unsigned short* __restrict__ attnb,  // [4096][2048] bf16
    const unsigned short* __restrict__ woutT,  // [2048][2048] bf16 (transposed)
    float* __restrict__ out) {
  __shared__ __align__(16) unsigned short Alds[2 * 64 * 32];
  __shared__ __align__(16) unsigned short Blds[2 * 128 * 32];

  const int mblk = blockIdx.x, nblk = blockIdx.y;
  v8f acc[8];
#pragma unroll
  for (int i = 0; i < 8; ++i) acc[i] = zero8();

  gemm_tile_64x128(attnb, woutT, 2048, 2048, mblk * 64, nblk * 128, 2048, Alds, Blds, acc);

  const int lane = threadIdx.x & 31, wave = threadIdx.x >> 5;
  const int mh = wave >> 1, nh = wave & 1;
  const int n16 = lane & 15, hf = lane >> 4;
#pragma unroll
  for (int mi = 0; mi < 2; ++mi)
#pragma unroll
    for (int j = 0; j < 4; ++j)
#pragma unroll
      for (int i = 0; i < 8; ++i) {
        int row = mblk * 64 + mh * 32 + mi * 16 + i + 8 * hf;
        int col = nblk * 128 + wave_tile(nh, j) * 16 + n16;
        out[(size_t)row * 2048 + col] = acc[mi * 4 + j][i];
      }
}

// ---------------------------------------------------------------------------
// Launch
// ---------------------------------------------------------------------------
extern "C" void kernel_launch(void* const* d_in, const int* in_sizes, int n_in,
                              void* d_out, int out_size, void* d_ws, size_t ws_size,
                              hipStream_t stream) {
  const float* x     = (const float*)d_in[0];  // [2,2048,2048]
  const float* w_qkv = (const float*)d_in[1];  // [2048,6144]
  const float* w_out = (const float*)d_in[2];  // [2048,2048]
  const float* s_qk  = (const float*)d_in[3];  // scalar
  float* out = (float*)d_out;

  char* ws = (char*)d_ws;
  const size_t MB = 1024 * 1024;
  unsigned short* xb     = (unsigned short*)(ws + 0);         // 16 MB
  unsigned short* wqkvT  = (unsigned short*)(ws + 16 * MB);   // 24 MB
  unsigned short* woutT  = (unsigned short*)(ws + 40 * MB);   // 8 MB
  unsigned short* qb     = (unsigned short*)(ws + 48 * MB);   // 16 MB
  unsigned short* kb     = (unsigned short*)(ws + 64 * MB);   // 16 MB
  unsigned short* vb     = (unsigned short*)(ws + 80 * MB);   // 16 MB
  unsigned short* attnb  = (unsigned short*)(ws + 96 * MB);   // 16 MB

  k_f2bf<<<4096, 256, 0, stream>>>(x, xb, 2 * 2048 * 2048);
  k_transpose_bf<<<8192, 256, 0, stream>>>(w_qkv, wqkvT, 2048, 6144);
  k_transpose_bf<<<4096, 256, 0, stream>>>(w_out, woutT, 2048, 2048);

  k_qkv<<<dim3(64, 48), 128, 0, stream>>>(xb, wqkvT, s_qk, qb, kb, vb);
  k_attn<<<dim3(32, 32), 128, 0, stream>>>(qb, kb, vb, attnb);
  k_outproj<<<dim3(64, 16), 128, 0, stream>>>(attnb, woutT, out);
}
